// MultiHeadSelfAttention_532575944955
// MI455X (gfx1250) — matrix operand order repaired and checked by reading, not compiled, not run on hardware
//
#include <hip/hip_runtime.h>
#include <hip/hip_bf16.h>

// ---------------------------------------------------------------------------
// Types for WMMA operands (gfx1250, wave32)
// ---------------------------------------------------------------------------
typedef __attribute__((ext_vector_type(8)))  float          v8f;
typedef __attribute__((ext_vector_type(8)))  unsigned short v8u;
typedef __attribute__((ext_vector_type(16))) unsigned short v16u;
typedef __attribute__((ext_vector_type(16))) __bf16         v16bf;
typedef __attribute__((ext_vector_type(4)))  int            v4i;

static __device__ __forceinline__ v16u cat16(v8u lo, v8u hi) {
  return __builtin_shufflevector(lo, hi, 0,1,2,3,4,5,6,7,8,9,10,11,12,13,14,15);
}

static __device__ __forceinline__ v16bf as_bf16(v16u u) {
  union { v16u u; v16bf b; } x; x.u = u; return x.b;
}

// D = A(16x32 bf16) x B(32x16 bf16) + C(16x16 f32)
static __device__ __forceinline__ v8f wmma_bf16(v16u a, v16u b, v8f c) {
  return __builtin_amdgcn_wmma_f32_16x16x32_bf16(
      /*neg_a=*/false, as_bf16(a), /*neg_b=*/false, as_bf16(b),
      /*c_mod=*/(short)0, c, /*reuse_a=*/false, /*reuse_b=*/false);
}

static __device__ __forceinline__ unsigned short f2bf(float f) {
  unsigned int u = __float_as_uint(f);
  unsigned int r = u + 0x7fffu + ((u >> 16) & 1u);
  return (unsigned short)(r >> 16);
}
static __device__ __forceinline__ float bf2f(unsigned short u) {
  return __uint_as_float(((unsigned int)u) << 16);
}

// ---------------------------------------------------------------------------
// Async global->LDS copy (gfx1250 async path). The builtin takes
// (v4i addrspace(1)*, v4i addrspace(3)*, imm offset, imm cpol).
// ---------------------------------------------------------------------------
#if defined(__has_builtin)
#  if __has_builtin(__builtin_amdgcn_global_load_async_to_lds_b128)
#    define HAS_ASYNC_LDS 1
#  endif
#endif
#ifndef HAS_ASYNC_LDS
#  define HAS_ASYNC_LDS 0
#endif

#if HAS_ASYNC_LDS
static __device__ __forceinline__ void async_cp16(const void* g, void* l) {
  __builtin_amdgcn_global_load_async_to_lds_b128(
      (__attribute__((address_space(1))) v4i*)(void*)g,
      (__attribute__((address_space(3))) v4i*)l,
      /*offset=*/0, /*cpol=*/0);
}
#endif

// ---------------------------------------------------------------------------
// Kernel 1: fp32 -> bf16 conversion
// ---------------------------------------------------------------------------
__global__ __launch_bounds__(256) void f32_to_bf16(const float* __restrict__ src,
                                                   unsigned short* __restrict__ dst,
                                                   int n) {
  int i = blockIdx.x * 256 + threadIdx.x;
  if (i < n) dst[i] = f2bf(src[i]);
}

// ---------------------------------------------------------------------------
// Kernel 2: GEMM  out[m, o] = sum_k A[m,k] * W[o,k]
//   A-panel (16 x K) staged in LDS (async copy when available), shared by all
//   8 waves; B streamed from global with double-buffered fragments.
//   mode 0: bf16 out, head-major  [B,H,S,64]   (Q, K pre-RoPE)
//   mode 1: bf16 out, head-transp [B,H,64,S]   (V^T, contiguous 16B stores)
//   mode 2: f32  out, row-major   [M,N]        (final projection)
// Grid = (M/16, N/512), block = 256.
// ---------------------------------------------------------------------------
__global__ __launch_bounds__(256) void gemm16_wmma(
    const unsigned short* __restrict__ A,
    const unsigned short* __restrict__ W,
    unsigned short* __restrict__ outB,
    float* __restrict__ outF,
    int M, int N, int K, int Sseq, int mode)
{
  __shared__ __attribute__((aligned(16))) unsigned short Atile[16 * 1024];

  const int lane = threadIdx.x & 31;
  const int wave = threadIdx.x >> 5;
  const int h16  = lane >> 4;
  const int l16  = lane & 15;
  const int mBase = blockIdx.x * 16;
  const int nBase = blockIdx.y * 512 + wave * 64;

  // ---- stage 16 x K A panel into LDS
  {
    const int chunksPerRow = K >> 3;               // 8 ushorts per 16B chunk
    for (int c = threadIdx.x; c < 16 * chunksPerRow; c += 256) {
      int row = c / chunksPerRow;
      int col = (c - row * chunksPerRow) << 3;
      const unsigned short* gp = A + (size_t)(mBase + row) * K + col;
      unsigned short*       lp = Atile + row * K + col;
#if HAS_ASYNC_LDS
      async_cp16(gp, lp);
#else
      *(v8u*)lp = *(const v8u*)gp;
#endif
    }
#if HAS_ASYNC_LDS
    asm volatile("s_wait_asynccnt 0" ::: "memory");
#endif
    __syncthreads();
  }

  // 32-bit lane offsets against uniform bases (SADDR + voffset form)
  unsigned boff[4];
#pragma unroll
  for (int nt = 0; nt < 4; ++nt)
    boff[nt] = (unsigned)(nBase + nt * 16 + l16) * (unsigned)K + h16 * 8;
  const unsigned short* Alds = Atile + l16 * K + h16 * 8;

  v8f zf = {0.f,0.f,0.f,0.f,0.f,0.f,0.f,0.f};
  v8f c[4] = { zf, zf, zf, zf };

  auto loadA = [&](int k0) -> v16u {
    return cat16(*(const v8u*)(Alds + k0), *(const v8u*)(Alds + k0 + 16));
  };
  auto loadB = [&](int k0, v16u bf[4]) {
#pragma unroll
    for (int nt = 0; nt < 4; ++nt) {
      const unsigned short* p = W + boff[nt] + (unsigned)k0;
      bf[nt] = cat16(*(const v8u*)p, *(const v8u*)(p + 16));
    }
  };

  // double-buffered main loop (K is a multiple of 64)
  v16u af0 = loadA(0), af1;
  v16u bf0[4], bf1[4];
  loadB(0, bf0);

  for (int k0 = 0; k0 < K; k0 += 64) {
    af1 = loadA(k0 + 32);
    loadB(k0 + 32, bf1);
#pragma unroll
    for (int nt = 0; nt < 4; ++nt) c[nt] = wmma_bf16(af0, bf0[nt], c[nt]);
    if (k0 + 64 < K) { af0 = loadA(k0 + 64); loadB(k0 + 64, bf0); }
#pragma unroll
    for (int nt = 0; nt < 4; ++nt) c[nt] = wmma_bf16(af1, bf1[nt], c[nt]);
  }

  // C/D layout: VGPR rr, lane -> row M = rr + 8*h16, col N = l16
  if (mode == 0) {
#pragma unroll
    for (int nt = 0; nt < 4; ++nt) {
      int o  = nBase + nt * 16 + l16;
      int hh = o >> 6, dh = o & 63;
#pragma unroll
      for (int rr = 0; rr < 8; ++rr) {
        int mrow = mBase + rr + 8 * h16;
        int bb = mrow / Sseq, ss = mrow % Sseq;
        outB[((size_t)(bb * 16 + hh) * Sseq + ss) * 64 + dh] = f2bf(c[nt][rr]);
      }
    }
  } else if (mode == 1) {
    int bb = mBase / Sseq;
    int sb = (mBase % Sseq) + 8 * h16;
#pragma unroll
    for (int nt = 0; nt < 4; ++nt) {
      int o  = nBase + nt * 16 + l16;
      int hh = o >> 6, dh = o & 63;
      v8u pk;
#pragma unroll
      for (int rr = 0; rr < 8; ++rr) pk[rr] = f2bf(c[nt][rr]);
      *(v8u*)(outB + ((size_t)(bb * 16 + hh) * 64 + dh) * Sseq + sb) = pk;
    }
  } else {
#pragma unroll
    for (int nt = 0; nt < 4; ++nt) {
      int o = nBase + nt * 16 + l16;
#pragma unroll
      for (int rr = 0; rr < 8; ++rr)
        outF[(size_t)(mBase + rr + 8 * h16) * N + o] = c[nt][rr];
    }
  }
}

// ---------------------------------------------------------------------------
// Kernel 3: in-place RoPE on [B,H,S,64] bf16.
// ---------------------------------------------------------------------------
__global__ __launch_bounds__(256) void rope_inplace(unsigned short* __restrict__ T,
                                                    const int* __restrict__ pos,
                                                    int Sseq, int H, int total)
{
  int t = blockIdx.x * 256 + threadIdx.x;
  if (t >= total) return;
  int i  = t & 31;
  int s  = (t >> 5) % Sseq;
  int bh = t / (32 * Sseq);
  int b  = bh / H;

  float p    = (float)pos[b * Sseq + s];
  float freq = __powf(10000.0f, -((float)(2 * i)) * (1.0f / 64.0f));
  float ang  = p * freq;
  float sn = __sinf(ang), cs = __cosf(ang);

  unsigned int* P32 = (unsigned int*)T;
  size_t idx = ((size_t)bh * Sseq + s) * 32 + i;
  unsigned int u = P32[idx];
  float ev = bf2f((unsigned short)(u & 0xffffu));
  float od = bf2f((unsigned short)(u >> 16));
  float oe = ev * cs - od * sn;
  float oo = ev * sn + od * cs;
  P32[idx] = (unsigned int)f2bf(oe) | ((unsigned int)f2bf(oo) << 16);
}

// ---------------------------------------------------------------------------
// Kernel 4: causal flash attention, one wave per (b,h, 16-query tile).
//   64-key blocks; mask-free full blocks + exactly one masked tail block.
//   Q,K: [B,H,S,64] bf16 (post-RoPE), V: [B,H,64,S] bf16 (transposed).
// ---------------------------------------------------------------------------
__global__ __launch_bounds__(256) void attn_wmma(
    const unsigned short* __restrict__ Q,
    const unsigned short* __restrict__ Kk,
    const unsigned short* __restrict__ Vt,
    unsigned short* __restrict__ outB,
    int Sseq, int qtPerBH, int H)
{
  __shared__ __attribute__((aligned(16))) unsigned short ptile[8][16 * 64];

  const int lane = threadIdx.x & 31;
  const int wib  = threadIdx.x >> 5;
  const int h16  = lane >> 4;
  const int l16  = lane & 15;

  const int w     = blockIdx.x * 8 + wib;
  const int qt    = w % qtPerBH;
  const int bh    = w / qtPerBH;
  const int qBase = qt * 16;

  const unsigned qkBase = (unsigned)bh * (unsigned)Sseq * 64u;
  const unsigned vBase  = qkBase;

  // Q fragments for both 32-wide K-dim steps of Dh=64
  v16u qf[2];
#pragma unroll
  for (int ds = 0; ds < 2; ++ds) {
    const unsigned short* p =
        Q + qkBase + (unsigned)(qBase + l16) * 64u + ds * 32 + h16 * 8;
    qf[ds] = cat16(*(const v8u*)p, *(const v8u*)(p + 16));
  }

  v8f zf = {0.f,0.f,0.f,0.f,0.f,0.f,0.f,0.f};
  v8f O[4] = { zf, zf, zf, zf };
  float mrow[8], lrow[8];
#pragma unroll
  for (int rr = 0; rr < 8; ++rr) { mrow[rr] = -1e30f; lrow[rr] = 0.f; }

  unsigned short* lds = &ptile[wib][0];

  // process one 64-key block: tiles kt in [0, ktn), optional causal mask
  auto block64 = [&](int kBase, int ktn, bool domask) {
    // ---- load all K fragments first (overlapped loads), then WMMA burst
    v16u kf[4][2];
#pragma unroll
    for (int kt = 0; kt < 4; ++kt) if (kt < ktn) {
#pragma unroll
      for (int ds = 0; ds < 2; ++ds) {
        const unsigned short* p = Kk + qkBase +
            (unsigned)(kBase + kt * 16 + l16) * 64u + ds * 32 + h16 * 8;
        kf[kt][ds] = cat16(*(const v8u*)p, *(const v8u*)(p + 16));
      }
    }
    v8f sc[4] = { zf, zf, zf, zf };
#pragma unroll
    for (int kt = 0; kt < 4; ++kt) if (kt < ktn) {
      sc[kt] = wmma_bf16(qf[0], kf[kt][0], sc[kt]);
      sc[kt] = wmma_bf16(qf[1], kf[kt][1], sc[kt]);
    }
    // ---- scale (+ causal mask only in tail block)
#pragma unroll
    for (int kt = 0; kt < 4; ++kt) if (kt < ktn) {
      int key = kBase + kt * 16 + l16;
#pragma unroll
      for (int rr = 0; rr < 8; ++rr) {
        float v = sc[kt][rr] * 0.125f;              // 1/sqrt(64)
        if (domask) {
          int qrow = qBase + rr + 8 * h16;
          v = (key > qrow) ? -1e30f : v;
        }
        sc[kt][rr] = v;
      }
    }
    // ---- online softmax, amortized over 64 keys
    float alpha[8];
#pragma unroll
    for (int rr = 0; rr < 8; ++rr) {
      float mx = -1e30f;
#pragma unroll
      for (int kt = 0; kt < 4; ++kt) if (kt < ktn) mx = fmaxf(mx, sc[kt][rr]);
      mx = fmaxf(mx, __shfl_xor(mx, 1, 32));
      mx = fmaxf(mx, __shfl_xor(mx, 2, 32));
      mx = fmaxf(mx, __shfl_xor(mx, 4, 32));
      mx = fmaxf(mx, __shfl_xor(mx, 8, 32));
      float mn = fmaxf(mrow[rr], mx);
      float rs = 0.f;
#pragma unroll
      for (int kt = 0; kt < 4; ++kt) if (kt < ktn) {
        float pv = __expf(sc[kt][rr] - mn);
        sc[kt][rr] = pv;
        rs += pv;
      }
      rs += __shfl_xor(rs, 1, 32);
      rs += __shfl_xor(rs, 2, 32);
      rs += __shfl_xor(rs, 4, 32);
      rs += __shfl_xor(rs, 8, 32);
      float a = __expf(mrow[rr] - mn);
      alpha[rr] = a;
      lrow[rr]  = lrow[rr] * a + rs;
      mrow[rr]  = mn;
    }
#pragma unroll
    for (int nt = 0; nt < 4; ++nt)
#pragma unroll
      for (int rr = 0; rr < 8; ++rr) O[nt][rr] *= alpha[rr];

    // ---- P: C-layout -> A-layout via per-wave LDS tile (row pitch 64)
#pragma unroll
    for (int kt = 0; kt < 4; ++kt) if (kt < ktn) {
      int col = kt * 16 + l16;
#pragma unroll
      for (int rr = 0; rr < 8; ++rr)
        lds[(rr + 8 * h16) * 64 + col] = f2bf(sc[kt][rr]);
    }
    if (ktn & 1) {                    // zero-fill odd partner tile for PV step
      int col = ktn * 16 + l16;
#pragma unroll
      for (int rr = 0; rr < 8; ++rr)
        lds[(rr + 8 * h16) * 64 + col] = 0;
    }
    asm volatile("s_wait_dscnt 0" ::: "memory");

    // ---- O += P @ V over ceil(ktn/2) 32-wide K-dim steps
    int steps = (ktn + 1) >> 1;
#pragma unroll
    for (int kd = 0; kd < 2; ++kd) if (kd < steps) {
      const unsigned short* pp = lds + l16 * 64 + kd * 32 + h16 * 8;
      v16u pf = cat16(*(const v8u*)pp, *(const v8u*)(pp + 16));
      v16u vf[4];
#pragma unroll
      for (int nt = 0; nt < 4; ++nt) {
        const unsigned short* p = Vt + vBase +
            (unsigned)(nt * 16 + l16) * (unsigned)Sseq + kBase + kd * 32 + h16 * 8;
        vf[nt] = cat16(*(const v8u*)p, *(const v8u*)(p + 16));
      }
#pragma unroll
      for (int nt = 0; nt < 4; ++nt) O[nt] = wmma_bf16(pf, vf[nt], O[nt]);
    }
  };

  // full (mask-free) 64-key blocks, then exactly one masked tail block
  const int E = ((qBase + 1) >> 6) << 6;
  for (int kb = 0; kb < E; kb += 64) block64(kb, 4, false);
  block64(E, ((qBase + 15 - E) >> 4) + 1, true);

  // ---- normalize and store attnOut[b, q, h*64 + d] (bf16)
  const int b  = bh / H;
  const int hh = bh % H;
#pragma unroll
  for (int nt = 0; nt < 4; ++nt) {
    int col = hh * 64 + nt * 16 + l16;
#pragma unroll
    for (int rr = 0; rr < 8; ++rr) {
      int qrow = qBase + rr + 8 * h16;
      outB[((size_t)(b * Sseq + qrow)) * 1024 + col] = f2bf(O[nt][rr] / lrow[rr]);
    }
  }
}

// ---------------------------------------------------------------------------
// Host-side pipeline
// ---------------------------------------------------------------------------
extern "C" void kernel_launch(void* const* d_in, const int* in_sizes, int n_in,
                              void* d_out, int out_size, void* d_ws, size_t ws_size,
                              hipStream_t stream) {
  (void)in_sizes; (void)n_in; (void)out_size; (void)ws_size;

  const float* x  = (const float*)d_in[0];
  const float* Wq = (const float*)d_in[1];
  const float* Wk = (const float*)d_in[2];
  const float* Wv = (const float*)d_in[3];
  const float* Wo = (const float*)d_in[4];
  const int*   tp = (const int*)d_in[5];

  const int B = 2, S = 2048, D = 1024, H = 16;
  const int M = B * S;

  unsigned short* ws   = (unsigned short*)d_ws;
  unsigned short* xb   = ws;                          // [M, D]
  unsigned short* Wqb  = xb  + (size_t)M * D;
  unsigned short* Wkb  = Wqb + (size_t)D * D;
  unsigned short* Wvb  = Wkb + (size_t)D * D;
  unsigned short* Wob  = Wvb + (size_t)D * D;
  unsigned short* Qh   = Wob + (size_t)D * D;         // [B,H,S,64]
  unsigned short* Kh   = Qh  + (size_t)M * D;         // [B,H,S,64]
  unsigned short* Vt   = Kh  + (size_t)M * D;         // [B,H,64,S]
  unsigned short* aOut = Vt  + (size_t)M * D;         // [B,S,D]

  f32_to_bf16<<<(M * D + 255) / 256, 256, 0, stream>>>(x,  xb,  M * D);
  f32_to_bf16<<<(D * D + 255) / 256, 256, 0, stream>>>(Wq, Wqb, D * D);
  f32_to_bf16<<<(D * D + 255) / 256, 256, 0, stream>>>(Wk, Wkb, D * D);
  f32_to_bf16<<<(D * D + 255) / 256, 256, 0, stream>>>(Wv, Wvb, D * D);
  f32_to_bf16<<<(D * D + 255) / 256, 256, 0, stream>>>(Wo, Wob, D * D);

  dim3 gg(M / 16, D / 512);
  gemm16_wmma<<<gg, 256, 0, stream>>>(xb, Wqb, Qh, nullptr, M, D, D, S, 0);
  gemm16_wmma<<<gg, 256, 0, stream>>>(xb, Wkb, Kh, nullptr, M, D, D, S, 0);
  gemm16_wmma<<<gg, 256, 0, stream>>>(xb, Wvb, Vt, nullptr, M, D, D, S, 1);

  int rtot = B * H * S * 32;
  rope_inplace<<<(rtot + 255) / 256, 256, 0, stream>>>(Qh, tp, S, H, rtot);
  rope_inplace<<<(rtot + 255) / 256, 256, 0, stream>>>(Kh, tp, S, H, rtot);

  int nWaves = B * H * (S / 16);
  attn_wmma<<<nWaves / 8, 256, 0, stream>>>(Qh, Kh, Vt, aOut, S, S / 16, H);

  gemm16_wmma<<<gg, 256, 0, stream>>>(aOut, Wob, nullptr, (float*)d_out,
                                      M, D, D, S, 2);
}
